// ArceeAttention_36240934043958
// MI455X (gfx1250) — matrix operand order repaired and checked by reading, not compiled, not run on hardware
//
#include <hip/hip_runtime.h>
#include <hip/hip_bf16.h>

// ---------------- problem constants ----------------
#define HIDDEN 4096
#define NH     32
#define NKV    8
#define HD     128
#define BB     2
#define SS     2048
#define TOKENS (BB*SS)          // 4096
#define QSZ    (NH*HD)          // 4096
#define KVSZ   (NKV*HD)         // 1024
#define QKV_N  (QSZ + 2*KVSZ)   // 6144
#define SCALE  0.08838834764831845f   // 1/sqrt(128)
#define LOG_THETA 9.210340371976184f  // ln(10000)

typedef __bf16 bf16;
typedef __attribute__((ext_vector_type(16))) __bf16 v16bf;
typedef __attribute__((ext_vector_type(8)))  __bf16 v8bf;
typedef __attribute__((ext_vector_type(4)))  __bf16 v4bf;
typedef __attribute__((ext_vector_type(8)))  float  v8f;
typedef __attribute__((ext_vector_type(4)))  float  v4f;
typedef __attribute__((ext_vector_type(4)))  unsigned u32x4;
typedef __attribute__((ext_vector_type(8)))  unsigned u32x8;

// ---------------- WMMA helpers ----------------
__device__ __forceinline__ v8f wmma_bf16(v16bf a, v16bf b, v8f c) {
  return __builtin_amdgcn_wmma_f32_16x16x32_bf16(false, a, false, b, (short)0, c,
                                                 false, false);
}

// A fragment (16x32): lane m = lane&15, g = lane>>4; K = 8g+e | 16+8g+(e-8)
__device__ __forceinline__ v16bf load_a_frag(const bf16* row, int g) {
  v8bf lo = *(const v8bf*)(row + 8 * g);
  v8bf hi = *(const v8bf*)(row + 16 + 8 * g);
  return __builtin_shufflevector(lo, hi, 0,1,2,3,4,5,6,7,8,9,10,11,12,13,14,15);
}
// B fragment (32x16): lane n = lane&15, g = lane>>4; K = 8g+e | 16+8g+(e-8)
__device__ __forceinline__ v16bf load_b_frag(const bf16* row, int g) {
  v8bf lo = *(const v8bf*)(row + 8 * g);
  v8bf hi = *(const v8bf*)(row + 16 + 8 * g);
  return __builtin_shufflevector(lo, hi, 0,1,2,3,4,5,6,7,8,9,10,11,12,13,14,15);
}

// ---------------- CDNA5 async / tensor data movement ----------------
// Per-lane 16B global -> LDS copy, tracked by ASYNCcnt (no VGPR round trip).
__device__ __forceinline__ void async_copy_b128(void* lptr, const void* gptr) {
  unsigned lds = (unsigned)(size_t)lptr;              // low 32 bits = LDS offset
  unsigned long long ga = (unsigned long long)(size_t)gptr;
  asm volatile("global_load_async_to_lds_b128 %0, %1, off"
               :: "v"(lds), "v"(ga) : "memory");
}
__device__ __forceinline__ void async_wait0() {
  asm volatile("s_wait_asynccnt 0" ::: "memory");
}

// TDM 2D tile load: copies tile_rows x (tile_units*8B) from a row-major tensor
// (row stride = row_stride_units*8B) into LDS at lds_off. data_size = 8 bytes.
__device__ __forceinline__ void tdm_load_2d(unsigned lds_off, const void* gaddr,
                                            unsigned row_units, unsigned nrows_tensor,
                                            unsigned tile_units, unsigned tile_rows,
                                            unsigned row_stride_units) {
  unsigned long long ga = (unsigned long long)(size_t)gaddr;
  u32x4 g0;
  g0[0] = 1u;                                           // count=1 (valid user D#)
  g0[1] = lds_off;                                      // lds_addr
  g0[2] = (unsigned)(ga & 0xFFFFFFFFu);                 // global_addr[31:0]
  g0[3] = (unsigned)((ga >> 32) & 0x1FFFFFFu) | (2u << 30);  // addr[56:32], type=2
  u32x8 g1;
  g1[0] = 3u << 16;                                     // wg_mask=0, data_size=8B
  g1[1] = (row_units & 0xFFFFu) << 16;                  // tensor_dim0[15:0]
  g1[2] = ((row_units >> 16) & 0xFFFFu) |               // tensor_dim0[31:16]
          ((nrows_tensor & 0xFFFFu) << 16);             // tensor_dim1[15:0]
  g1[3] = ((nrows_tensor >> 16) & 0xFFFFu) |            // tensor_dim1[31:16]
          ((tile_units & 0xFFFFu) << 16);               // tile_dim0
  g1[4] = (tile_rows & 0xFFFFu);                        // tile_dim1 (tile_dim2=0)
  g1[5] = row_stride_units;                             // tensor_dim0_stride[31:0]
  g1[6] = 0u;                                           // stride hi / dim1_stride
  g1[7] = 0u;
  asm volatile("tensor_load_to_lds %0, %1" :: "s"(g0), "s"(g1) : "memory");
}

// ---------------- f32 -> bf16 convert ----------------
__global__ __launch_bounds__(256) void f32_to_bf16_kernel(
    const float* __restrict__ in, bf16* __restrict__ out, long long n4) {
  long long i = (long long)blockIdx.x * 256 + threadIdx.x;
  long long stride = (long long)gridDim.x * 256;
  for (; i < n4; i += stride) {
    v4f x = ((const v4f*)in)[i];
    v4bf y;
#pragma unroll
    for (int e = 0; e < 4; ++e) y[e] = (bf16)x[e];
    ((v4bf*)out)[i] = y;
  }
}

// ---------------- bf16 GEMM, f32 accumulate ----------------
#define TM 128
#define TN 128
#define TK 32
__global__ __launch_bounds__(256) void gemm_bf16_wmma(
    const bf16* __restrict__ A, const bf16* __restrict__ B,
    float* __restrict__ Cf, bf16* __restrict__ Cb,
    int M, int N, int K, int bf_out) {
  __shared__ bf16 As[TM * TK];   // [m][k]
  __shared__ bf16 Bs[TN * TK];   // [n][k] (transposed)
  const int t = threadIdx.x;
  const int lane = t & 31;
  const int wave = t >> 5;
  const int g  = lane >> 4;
  const int ln = lane & 15;
  const int m0 = blockIdx.y * TM;
  const int n0 = blockIdx.x * TN;
  const int wm = (wave >> 1) * 32;
  const int wn = (wave & 1) * 64;

  v8f acc[2][4];
#pragma unroll
  for (int i = 0; i < 2; ++i)
#pragma unroll
    for (int j = 0; j < 4; ++j) acc[i][j] = (v8f){};

  for (int k0 = 0; k0 < K; k0 += TK) {
    __syncthreads();
    // A tile 128x32: verbatim copy -> async global->LDS DMA (ASYNCcnt)
#pragma unroll
    for (int it = 0; it < 2; ++it) {
      int idx = t + it * 256;
      int row = idx >> 2;
      int seg = idx & 3;
      async_copy_b128(&As[row * TK + seg * 8],
                      &A[(size_t)(m0 + row) * K + k0 + seg * 8]);
    }
    // B tile 32x128: needs transpose. Issue BOTH global loads into distinct
    // registers first, then do the LDS scatter stores (keeps 2 loads in flight
    // instead of load->wait->store serialization).
    v8bf vb[2];
#pragma unroll
    for (int it = 0; it < 2; ++it) {
      int idx = t + it * 256;
      int krow = idx >> 4;
      int seg  = idx & 15;
      vb[it] = *(const v8bf*)&B[(size_t)(k0 + krow) * N + n0 + seg * 8];
    }
#pragma unroll
    for (int it = 0; it < 2; ++it) {
      int idx = t + it * 256;
      int krow = idx >> 4;
      int seg  = idx & 15;
#pragma unroll
      for (int e = 0; e < 8; ++e) Bs[(seg * 8 + e) * TK + krow] = vb[it][e];
    }
    if (k0 + TK < K) {  // global_prefetch_b8 for next tiles
      __builtin_prefetch(&A[(size_t)(m0 + (t >> 1)) * K + k0 + TK], 0, 0);
      __builtin_prefetch(&B[(size_t)(k0 + TK + (t >> 3)) * N + n0 + (t & 7) * 16], 0, 0);
    }
    async_wait0();
    __syncthreads();

    v16bf a0 = load_a_frag(&As[(wm + ln) * TK], g);
    v16bf a1 = load_a_frag(&As[(wm + 16 + ln) * TK], g);
#pragma unroll
    for (int j = 0; j < 4; ++j) {
      v16bf b = load_b_frag(&Bs[(wn + j * 16 + ln) * TK], g);
      acc[0][j] = wmma_bf16(a0, b, acc[0][j]);
      acc[1][j] = wmma_bf16(a1, b, acc[1][j]);
    }
  }

#pragma unroll
  for (int i = 0; i < 2; ++i)
#pragma unroll
    for (int j = 0; j < 4; ++j)
#pragma unroll
      for (int v = 0; v < 8; ++v) {
        int row = m0 + wm + i * 16 + v + 8 * g;
        int col = n0 + wn + j * 16 + ln;
        float val = acc[i][j][v];
        if (bf_out) Cb[(size_t)row * N + col] = (bf16)val;
        else        Cf[(size_t)row * N + col] = val;
      }
}

// ---------------- RoPE (NeoX) + layout to [b][h][s][d] ----------------
__global__ __launch_bounds__(256) void rope_kernel(
    const bf16* __restrict__ qkv, const int* __restrict__ pos,
    bf16* __restrict__ Qb, bf16* __restrict__ Kb, bf16* __restrict__ Vb) {
  int tkn = blockIdx.x;
  int b = tkn / SS, s = tkn % SS;
  float p = (float)pos[tkn];
  const bf16* base = qkv + (size_t)tkn * QKV_N;

  for (int idx = threadIdx.x; idx < NH * 64; idx += 256) {
    int hh = idx >> 6, i = idx & 63;
    float inv = __expf(-(float)i * (LOG_THETA / 64.0f));
    float ang = p * inv;
    float sn, cs;
    __sincosf(ang, &sn, &cs);
    float x1 = (float)base[hh * HD + i];
    float x2 = (float)base[hh * HD + i + 64];
    size_t o = (((size_t)b * NH + hh) * SS + s) * HD;
    Qb[o + i]      = (bf16)(x1 * cs - x2 * sn);
    Qb[o + i + 64] = (bf16)(x2 * cs + x1 * sn);
  }
  for (int idx = threadIdx.x; idx < NKV * 64; idx += 256) {
    int hh = idx >> 6, i = idx & 63;
    float inv = __expf(-(float)i * (LOG_THETA / 64.0f));
    float ang = p * inv;
    float sn, cs;
    __sincosf(ang, &sn, &cs);
    float x1 = (float)base[QSZ + hh * HD + i];
    float x2 = (float)base[QSZ + hh * HD + i + 64];
    size_t o = (((size_t)b * NKV + hh) * SS + s) * HD;
    Kb[o + i]      = (bf16)(x1 * cs - x2 * sn);
    Kb[o + i + 64] = (bf16)(x2 * cs + x1 * sn);
  }
  for (int idx = threadIdx.x; idx < NKV * HD; idx += 256) {
    int hh = idx >> 7, i = idx & 127;
    Vb[(((size_t)b * NKV + hh) * SS + s) * HD + i] = base[QSZ + KVSZ + hh * HD + i];
  }
}

// ---------------- flash attention (causal, GQA) ----------------
#define CK 32
__global__ __launch_bounds__(256) void attn_kernel(
    const bf16* __restrict__ Qb, const bf16* __restrict__ Kb,
    const bf16* __restrict__ Vb, bf16* __restrict__ Ob) {
  __shared__ bf16 lk[CK * HD];        // K chunk [kv][d] (B layout for Q*K^T)
  __shared__ bf16 lvt[HD * CK];       // V chunk transposed [d][kv]
  __shared__ bf16 lp[8 * 16 * CK];    // per-wave P tile [16][32]

  const int t = threadIdx.x;
  const int lane = t & 31, wave = t >> 5;
  const int g = lane >> 4, ln = lane & 15;
  const int qblk = blockIdx.x;
  const int h    = blockIdx.y;
  const int b    = blockIdx.z;
  const int kvh  = h / (NH / NKV);
  const int q0   = qblk * 128;
  const int wq   = q0 + wave * 16;

  const bf16* qptr = Qb + ((size_t)b * NH + h) * SS * HD;
  const bf16* kptr = Kb + ((size_t)b * NKV + kvh) * SS * HD;
  const bf16* vptr = Vb + ((size_t)b * NKV + kvh) * SS * HD;

  v16bf qf[4];
#pragma unroll
  for (int c = 0; c < 4; ++c)
    qf[c] = load_a_frag(qptr + (size_t)(wq + ln) * HD + c * 32, g);

  v8f acc[8];
#pragma unroll
  for (int j = 0; j < 8; ++j) acc[j] = (v8f){};
  float rm[8], rs[8], alpha[8];
#pragma unroll
  for (int v = 0; v < 8; ++v) { rm[v] = -1e30f; rs[v] = 0.0f; }

  const int kv_end = q0 + 128;
  for (int kv0 = 0; kv0 < kv_end; kv0 += CK) {
    __syncthreads();
    // K chunk: pure 2D tile copy -> TDM (one DMA per chunk, wave 0 issues).
    if (t < 32) {
      tdm_load_2d((unsigned)(size_t)&lk[0], &kptr[(size_t)kv0 * HD],
                  (HD * 2) / 8, SS, (HD * 2) / 8, CK, (HD * 2) / 8);
    }
    // V chunk transposed [128][32]: manual (TDM cannot transpose).
    // Both global loads issued before any LDS store.
    v8bf vv[2];
#pragma unroll
    for (int it = 0; it < 2; ++it) {
      int idx = t + 256 * it;
      int row = idx >> 4, seg = idx & 15;
      vv[it] = *(const v8bf*)&vptr[(size_t)(kv0 + row) * HD + seg * 8];
    }
#pragma unroll
    for (int it = 0; it < 2; ++it) {
      int idx = t + 256 * it;
      int row = idx >> 4, seg = idx & 15;
#pragma unroll
      for (int e = 0; e < 8; ++e) lvt[(seg * 8 + e) * CK + row] = vv[it][e];
    }
    if (t < 32) __builtin_amdgcn_s_wait_tensorcnt(0);
    __syncthreads();

    v8f sc0 = (v8f){}, sc1 = (v8f){};
#pragma unroll
    for (int c = 0; c < 4; ++c) {
      v16bf kb0 = load_b_frag(&lk[(ln) * HD + c * 32], g);
      v16bf kb1 = load_b_frag(&lk[(16 + ln) * HD + c * 32], g);
      sc0 = wmma_bf16(qf[c], kb0, sc0);
      sc1 = wmma_bf16(qf[c], kb1, sc1);
    }

    float p0v[8], p1v[8];
#pragma unroll
    for (int v = 0; v < 8; ++v) {
      int qg = wq + v + 8 * g;
      float s0 = sc0[v] * SCALE; if (kv0 + ln > qg)      s0 = -1e30f;
      float s1 = sc1[v] * SCALE; if (kv0 + 16 + ln > qg) s1 = -1e30f;
      float mx = fmaxf(s0, s1);
      mx = fmaxf(mx, __shfl_xor(mx, 1, 32));
      mx = fmaxf(mx, __shfl_xor(mx, 2, 32));
      mx = fmaxf(mx, __shfl_xor(mx, 4, 32));
      mx = fmaxf(mx, __shfl_xor(mx, 8, 32));
      float mold = rm[v];
      float mn = fmaxf(mold, mx);
      rm[v] = mn;
      float al = __expf(mold - mn);
      float p0 = __expf(s0 - mn);
      float p1 = __expf(s1 - mn);
      float ps = p0 + p1;
      ps += __shfl_xor(ps, 1, 32);
      ps += __shfl_xor(ps, 2, 32);
      ps += __shfl_xor(ps, 4, 32);
      ps += __shfl_xor(ps, 8, 32);
      rs[v] = rs[v] * al + ps;
      alpha[v] = al;
      p0v[v] = p0; p1v[v] = p1;
    }

#pragma unroll
    for (int j = 0; j < 8; ++j)
#pragma unroll
      for (int v = 0; v < 8; ++v) acc[j][v] *= alpha[v];

    bf16* pw = &lp[wave * 16 * CK];
#pragma unroll
    for (int v = 0; v < 8; ++v) {
      pw[(v + 8 * g) * CK + ln]      = (bf16)p0v[v];
      pw[(v + 8 * g) * CK + 16 + ln] = (bf16)p1v[v];
    }
    v16bf pf = load_a_frag(&pw[ln * CK], g);

#pragma unroll
    for (int j = 0; j < 8; ++j) {
      v16bf vf = load_b_frag(&lvt[(j * 16 + ln) * CK], g);
      acc[j] = wmma_bf16(pf, vf, acc[j]);
    }
  }

#pragma unroll
  for (int j = 0; j < 8; ++j)
#pragma unroll
    for (int v = 0; v < 8; ++v) {
      int qrow = wq + v + 8 * g;
      float val = acc[j][v] / rs[v];
      Ob[((size_t)b * SS + qrow) * QSZ + h * HD + j * 16 + ln] = (bf16)val;
    }
}

// ---------------- host launch ----------------
extern "C" void kernel_launch(void* const* d_in, const int* in_sizes, int n_in,
                              void* d_out, int out_size, void* d_ws, size_t ws_size,
                              hipStream_t stream) {
  (void)in_sizes; (void)n_in; (void)out_size; (void)ws_size;
  const float* hidden    = (const float*)d_in[0];
  const int*   positions = (const int*)d_in[1];
  const float* w_qkv     = (const float*)d_in[2];
  const float* w_o       = (const float*)d_in[3];
  float* out = (float*)d_out;

  char* ws = (char*)d_ws;
  size_t off = 0;
  auto take = [&](size_t bytes) -> char* {
    char* p = ws + off;
    off += (bytes + 255) & ~(size_t)255;
    return p;
  };
  bf16* hid_bf  = (bf16*)take((size_t)TOKENS * HIDDEN * 2);
  bf16* wqkv_bf = (bf16*)take((size_t)HIDDEN * QKV_N * 2);
  bf16* wo_bf   = (bf16*)take((size_t)QSZ * HIDDEN * 2);
  bf16* qkv_bf  = (bf16*)take((size_t)TOKENS * QKV_N * 2);
  bf16* q_bf    = (bf16*)take((size_t)TOKENS * QSZ * 2);
  bf16* k_bf    = (bf16*)take((size_t)TOKENS * KVSZ * 2);
  bf16* v_bf    = (bf16*)take((size_t)TOKENS * KVSZ * 2);
  bf16* attn_bf = (bf16*)take((size_t)TOKENS * QSZ * 2);

  auto cvt = [&](const float* src, bf16* dst, size_t n) {
    long long n4 = (long long)(n / 4);
    long long blocks = (n4 + 255) / 256;
    if (blocks > 8192) blocks = 8192;
    f32_to_bf16_kernel<<<(int)blocks, 256, 0, stream>>>(src, dst, n4);
  };
  cvt(hidden, hid_bf, (size_t)TOKENS * HIDDEN);
  cvt(w_qkv,  wqkv_bf, (size_t)HIDDEN * QKV_N);
  cvt(w_o,    wo_bf,   (size_t)QSZ * HIDDEN);

  gemm_bf16_wmma<<<dim3(QKV_N / TN, TOKENS / TM), 256, 0, stream>>>(
      hid_bf, wqkv_bf, nullptr, qkv_bf, TOKENS, QKV_N, HIDDEN, 1);

  rope_kernel<<<TOKENS, 256, 0, stream>>>(qkv_bf, positions, q_bf, k_bf, v_bf);

  attn_kernel<<<dim3(SS / 128, NH, BB), 256, 0, stream>>>(q_bf, k_bf, v_bf, attn_bf);

  gemm_bf16_wmma<<<dim3(HIDDEN / TN, TOKENS / TM), 256, 0, stream>>>(
      attn_bf, wo_bf, out, nullptr, TOKENS, HIDDEN, QSZ, 0);
}
